// BeamSeq2Seq_56092272886282
// MI455X (gfx1250) — matrix operand order repaired and checked by reading, not compiled, not run on hardware
//
#include <hip/hip_runtime.h>
#include <hip/hip_bf16.h>
#include <math.h>

// ---------------------------------------------------------------------------
// Beam-search seq2seq decoder for MI455X (gfx1250, wave32, WMMA bf16).
// Dominant cost: per-step logits GEMM (16 x 1024) @ Wc^T (1024 x 50257).
// Strategy: one-time fp32->bf16 weight conversion into workspace (Wc_bf16 =
// 103 MB -> L2-resident on the 192MB L2), then every GEMM runs on
// V_WMMA_F32_16X16X32_BF16 tiles. 64 sequential steps, ~9 kernels each.
// ---------------------------------------------------------------------------

#define VOCAB   50257
#define VTILES  3144            // ceil(50257/16) rounded to 8-wave blocks
#define VPAD    (VTILES * 16)   // 50304
#define EMB     512
#define HID     512
#define QDIM    512
#define TSTEPS  2048
#define BEAM    16
#define MAXLEN  64
#define SOS_TOK 1
#define EOS_TOK 2
#define NEGF    (-1e9f)

typedef __bf16 bf16_t;
typedef __attribute__((ext_vector_type(16))) __bf16 bf16x16;
typedef __attribute__((ext_vector_type(8)))  __bf16 bf16x8;
typedef __attribute__((ext_vector_type(8)))  float  f32x8;

__device__ __forceinline__ float sigf(float x) { return 1.0f / (1.0f + __expf(-x)); }

// ---- WMMA fragment helpers (wave32; A/B in contiguous k order, not the ISA table; C/D per CDNA5 ISA) ----
// A (16x32 bf16, row-major source, stride lda elements):
//   lane l: M = l&15 ; lanes 0-15 hold K 0..15, lanes 16-31 K 16..31
__device__ __forceinline__ bf16x16 load_a_frag(const bf16_t* A, int lda, int k0, int lane) {
    int m  = lane & 15;
    int kb = (lane < 16) ? 0 : 16;
    const bf16_t* p0 = A + (size_t)m * lda + k0 + kb;
    bf16x8 lo = *(const bf16x8*)(p0);
    bf16x8 hi = *(const bf16x8*)(p0 + 8);
    bf16x16 r;
#pragma unroll
    for (int i = 0; i < 8; ++i) { r[i] = lo[i]; r[i + 8] = hi[i]; }
    return r;
}
// B (32x16 bf16). Weights stored K-contiguous per output column n (= PyTorch
// (N,K) row-major). lane l: N = n0+(l&15); lanes 0-15 K=0..15, 16-31 K=16..31.
__device__ __forceinline__ bf16x16 load_b_frag(const bf16_t* B, int ldb, int n0, int k0, int lane) {
    int n    = n0 + (lane & 15);
    int koff = (lane < 16) ? 0 : 16;
    return *(const bf16x16*)(B + (size_t)n * ldb + k0 + koff);
}
__device__ __forceinline__ f32x8 wmma_bf16(bf16x16 a, bf16x16 b, f32x8 c) {
    return __builtin_amdgcn_wmma_f32_16x16x32_bf16(false, a, false, b, (short)0, c, false, false);
}
// D/C 16x16 f32: lane l col = n0+(l&15); vgpr j -> row (l<16 ? j : j+8)
__device__ __forceinline__ void store_d(float* D, int ldd, int n0, f32x8 acc, int lane) {
    int col = n0 + (lane & 15);
    int mb  = (lane < 16) ? 0 : 8;
#pragma unroll
    for (int j = 0; j < 8; ++j) D[(size_t)(mb + j) * ldd + col] = acc[j];
}

// ---------------------------- conversion kernels ----------------------------
__global__ void k_cvt(const float* __restrict__ s, bf16_t* __restrict__ d, size_t n) {
    size_t i = (size_t)blockIdx.x * blockDim.x + threadIdx.x;
    size_t st = (size_t)gridDim.x * blockDim.x;
    for (; i < n; i += st) d[i] = (bf16_t)s[i];
}
// dst[n][0:Ka] = A[n][:], dst[n][Ka:Ka+Kb] = B[n][:]   (concat along K)
__global__ void k_cvt_cat(const float* __restrict__ A, const float* __restrict__ B,
                          bf16_t* __restrict__ d, int N, int Ka, int Kb) {
    int K = Ka + Kb;
    size_t n = (size_t)N * K;
    size_t i = (size_t)blockIdx.x * blockDim.x + threadIdx.x;
    size_t st = (size_t)gridDim.x * blockDim.x;
    for (; i < n; i += st) {
        int r = (int)(i / K), k = (int)(i % K);
        d[i] = (bf16_t)(k < Ka ? A[(size_t)r * Ka + k] : B[(size_t)r * Kb + (k - Ka)]);
    }
}
// valT[q][t] = value[t][q]   (512 x 2048 from 2048 x 512)
__global__ void k_cvt_valT(const float* __restrict__ v, bf16_t* __restrict__ d) {
    size_t n = (size_t)QDIM * TSTEPS;
    size_t i = (size_t)blockIdx.x * blockDim.x + threadIdx.x;
    size_t st = (size_t)gridDim.x * blockDim.x;
    for (; i < n; i += st) {
        int q = (int)(i / TSTEPS), t = (int)(i % TSTEPS);
        d[i] = (bf16_t)v[(size_t)t * QDIM + q];
    }
}

// ------------------------------ state init ---------------------------------
__global__ void k_init(float* h1, float* c1, float* h2, float* c2, float* ctx,
                       float* scores, int* prev, int* finished, int* lengths, int* seqs) {
    int i = blockIdx.x * 256 + threadIdx.x;
    if (i < BEAM * HID) { h1[i] = 0; c1[i] = 0; h2[i] = 0; c2[i] = 0; ctx[i] = 0; }
    if (i < BEAM) { scores[i] = 0; prev[i] = SOS_TOK; finished[i] = 0; lengths[i] = 0; }
    if (i < BEAM * (MAXLEN + 1)) { seqs[i] = ((i % (MAXLEN + 1)) == 0) ? SOS_TOK : EOS_TOK; }
}

// ---------------- LSTM cells: gates via WMMA, pointwise in LDS -------------
// Block = 128 threads = 4 waves; wave w computes gate tile (i,f,g,o) for the
// 16 hidden columns of this block; K streamed from bf16 weights.
__global__ __launch_bounds__(128) void k_lstm1(
    const float* __restrict__ emb, const int* __restrict__ prev,
    const float* __restrict__ ctx, const float* __restrict__ h1c, const float* __restrict__ c1c,
    const bf16_t* __restrict__ B1, const float* __restrict__ bih, const float* __restrict__ bhh,
    float* __restrict__ h1n, float* __restrict__ c1n) {
    __shared__ __attribute__((aligned(64))) bf16_t A1[BEAM * 1536];
    __shared__ float gates[4][16][16];
    const int tid = threadIdx.x;
    for (int i = tid; i < BEAM * 1536; i += 128) {
        int m = i / 1536, c = i % 1536;
        float v;
        if (c < 512)       v = emb[(size_t)prev[m] * EMB + c];
        else if (c < 1024) v = ctx[m * QDIM + (c - 512)];
        else               v = h1c[m * HID + (c - 1024)];
        A1[i] = (bf16_t)v;
    }
    __syncthreads();
    const int lane = tid & 31, wave = tid >> 5;
    const int n0 = blockIdx.x * 16;
    const int ng = wave * HID + n0;                 // gate-major output column
    f32x8 acc = {};
    for (int k0 = 0; k0 < 1536; k0 += 32)
        acc = wmma_bf16(load_a_frag(A1, 1536, k0, lane), load_b_frag(B1, 1536, ng, k0, lane), acc);
    {
        int col = lane & 15, mb = (lane < 16) ? 0 : 8;
        float bias = bih[ng + col] + bhh[ng + col];
#pragma unroll
        for (int j = 0; j < 8; ++j) gates[wave][mb + j][col] = acc[j] + bias;
    }
    __syncthreads();
    for (int i = tid; i < 256; i += 128) {
        int m = i >> 4, col = i & 15;
        float gi = gates[0][m][col], gf = gates[1][m][col];
        float gg = gates[2][m][col], go = gates[3][m][col];
        float cold = c1c[m * HID + n0 + col];
        float cn = sigf(gf) * cold + sigf(gi) * tanhf(gg);
        c1n[m * HID + n0 + col] = cn;
        h1n[m * HID + n0 + col] = sigf(go) * tanhf(cn);
    }
}

__global__ __launch_bounds__(128) void k_lstm2(
    const float* __restrict__ h1n, const float* __restrict__ h2c, const float* __restrict__ c2c,
    const bf16_t* __restrict__ B2, const float* __restrict__ bih, const float* __restrict__ bhh,
    float* __restrict__ h2n, float* __restrict__ c2n) {
    __shared__ __attribute__((aligned(64))) bf16_t A2[BEAM * 1024];
    __shared__ float gates[4][16][16];
    const int tid = threadIdx.x;
    for (int i = tid; i < BEAM * 1024; i += 128) {
        int m = i >> 10, c = i & 1023;
        A2[i] = (bf16_t)(c < 512 ? h1n[m * HID + c] : h2c[m * HID + (c - 512)]);
    }
    __syncthreads();
    const int lane = tid & 31, wave = tid >> 5;
    const int n0 = blockIdx.x * 16;
    const int ng = wave * HID + n0;
    f32x8 acc = {};
    for (int k0 = 0; k0 < 1024; k0 += 32)
        acc = wmma_bf16(load_a_frag(A2, 1024, k0, lane), load_b_frag(B2, 1024, ng, k0, lane), acc);
    {
        int col = lane & 15, mb = (lane < 16) ? 0 : 8;
        float bias = bih[ng + col] + bhh[ng + col];
#pragma unroll
        for (int j = 0; j < 8; ++j) gates[wave][mb + j][col] = acc[j] + bias;
    }
    __syncthreads();
    for (int i = tid; i < 256; i += 128) {
        int m = i >> 4, col = i & 15;
        float gi = gates[0][m][col], gf = gates[1][m][col];
        float gg = gates[2][m][col], go = gates[3][m][col];
        float cold = c2c[m * HID + n0 + col];
        float cn = sigf(gf) * cold + sigf(gi) * tanhf(gg);
        c2n[m * HID + n0 + col] = cn;
        h2n[m * HID + n0 + col] = sigf(go) * tanhf(cn);
    }
}

// --------- query = h2 @ Wq^T + bq, then energy = query . key  --------------
// 16 blocks x 8 waves. Each block redundantly computes the (cheap) query into
// LDS bf16, then each wave does one 16-wide T tile of the energy GEMM.
__global__ __launch_bounds__(256) void k_attn_energy(
    const float* __restrict__ h2, const bf16_t* __restrict__ BWq, const float* __restrict__ bq,
    const bf16_t* __restrict__ Bkey, float* __restrict__ energy) {
    __shared__ __attribute__((aligned(64))) bf16_t Ah[BEAM * QDIM];
    __shared__ __attribute__((aligned(64))) bf16_t Aq[BEAM * QDIM];
    const int tid = threadIdx.x;
    for (int i = tid; i < BEAM * QDIM; i += 256) Ah[i] = (bf16_t)h2[i];
    __syncthreads();
    const int lane = tid & 31, wave = tid >> 5;
    for (int j = 0; j < 4; ++j) {                   // 8 waves x 4 = 32 query tiles
        int n0 = (wave * 4 + j) * 16;
        f32x8 acc = {};
        for (int k0 = 0; k0 < QDIM; k0 += 32)
            acc = wmma_bf16(load_a_frag(Ah, QDIM, k0, lane), load_b_frag(BWq, QDIM, n0, k0, lane), acc);
        int col = n0 + (lane & 15), mb = (lane < 16) ? 0 : 8;
        float bias = bq[col];
#pragma unroll
        for (int jj = 0; jj < 8; ++jj) Aq[(mb + jj) * QDIM + col] = (bf16_t)(acc[jj] + bias);
    }
    __syncthreads();
    int t0 = blockIdx.x * 128 + wave * 16;          // 16 blocks x 8 waves = 2048 cols
    f32x8 acc = {};
    for (int k0 = 0; k0 < QDIM; k0 += 32)
        acc = wmma_bf16(load_a_frag(Aq, QDIM, k0, lane), load_b_frag(Bkey, QDIM, t0, k0, lane), acc);
    store_d(energy, TSTEPS, t0, acc, lane);
}

// ------------- softmax + masked renorm, emit attn weights as bf16 ----------
__global__ __launch_bounds__(256) void k_softmax(
    const float* __restrict__ energy, const unsigned char* __restrict__ mask,
    bf16_t* __restrict__ attn) {
    __shared__ float red[256];
    const int r = blockIdx.x, t = threadIdx.x;
    const float* e = energy + (size_t)r * TSTEPS;
    float mx = -3.4e38f;
    for (int i = t; i < TSTEPS; i += 256) mx = fmaxf(mx, e[i]);
    red[t] = mx; __syncthreads();
    for (int s = 128; s > 0; s >>= 1) { if (t < s) red[t] = fmaxf(red[t], red[t + s]); __syncthreads(); }
    mx = red[0]; __syncthreads();
    float sm = 0.f;
    for (int i = t; i < TSTEPS; i += 256) sm += __expf(e[i] - mx);
    red[t] = sm; __syncthreads();
    for (int s = 128; s > 0; s >>= 1) { if (t < s) red[t] += red[t + s]; __syncthreads(); }
    const float denom = red[0]; __syncthreads();
    float sm2 = 0.f;
    for (int i = t; i < TSTEPS; i += 256) {
        float a = __expf(e[i] - mx) / denom;
        if (!mask[i]) a = 0.f;
        sm2 += a;
    }
    red[t] = sm2; __syncthreads();
    for (int s = 128; s > 0; s >>= 1) { if (t < s) red[t] += red[t + s]; __syncthreads(); }
    const float d2 = fmaxf(red[0], 2e-30f);
    for (int i = t; i < TSTEPS; i += 256) {
        float a = __expf(e[i] - mx) / denom;
        if (!mask[i]) a = 0.f;
        attn[(size_t)r * TSTEPS + i] = (bf16_t)(a / d2);
    }
}

// ------------------ context = attn (16x2048) @ valT^T  ---------------------
__global__ __launch_bounds__(256) void k_context(
    const bf16_t* __restrict__ attn, const bf16_t* __restrict__ BvalT, float* __restrict__ ctxn) {
    const int lane = threadIdx.x & 31, wave = threadIdx.x >> 5;
    const int q0 = (blockIdx.x * 8 + wave) * 16;    // 4 blocks x 8 waves = 32 tiles
    f32x8 acc = {};
    for (int k0 = 0; k0 < TSTEPS; k0 += 32)
        acc = wmma_bf16(load_a_frag(attn, TSTEPS, k0, lane), load_b_frag(BvalT, TSTEPS, q0, k0, lane), acc);
    store_d(ctxn, QDIM, q0, acc, lane);
}

// ------------- logits = [h2|ctx] (16x1024) @ Wc^T (L2-resident bf16) -------
__global__ __launch_bounds__(256) void k_logits(
    const float* __restrict__ h2, const float* __restrict__ ctx,
    const bf16_t* __restrict__ BWc, const float* __restrict__ bc, float* __restrict__ logits) {
    __shared__ __attribute__((aligned(64))) bf16_t A[BEAM * 1024];
    const int tid = threadIdx.x;
    for (int i = tid; i < BEAM * 1024; i += 256) {
        int m = i >> 10, c = i & 1023;
        A[i] = (bf16_t)(c < 512 ? h2[m * HID + c] : ctx[m * QDIM + (c - 512)]);
    }
    __syncthreads();
    const int lane = tid & 31, wave = tid >> 5;
    const int v0 = (blockIdx.x * 8 + wave) * 16;    // 393 blocks x 8 waves = 3144 tiles
    int n = v0 + (lane & 15);
    int nsafe = (n < VOCAB) ? n : (VOCAB - 1);      // keep EXEC uniform; fix on store
    const bf16_t* brow = BWc + (size_t)nsafe * 1024;
    const int koff = (lane < 16) ? 0 : 16;
    f32x8 acc = {};
    for (int k0 = 0; k0 < 1024; k0 += 32) {
        bf16x16 b = *(const bf16x16*)(brow + k0 + koff);
        acc = wmma_bf16(load_a_frag(A, 1024, k0, lane), b, acc);
    }
    const int col = v0 + (lane & 15), mb = (lane < 16) ? 0 : 8;
    const float bias = (col < VOCAB) ? bc[col] : 0.f;
#pragma unroll
    for (int j = 0; j < 8; ++j)
        logits[(size_t)(mb + j) * VPAD + col] = (col < VOCAB) ? (acc[j] + bias) : NEGF;
}

// ------------------------- per-row max / logsumexp -------------------------
__global__ __launch_bounds__(256) void k_rowstats(
    const float* __restrict__ logits, float* __restrict__ rowmax, float* __restrict__ rowlse) {
    __shared__ float red[256];
    const int r = blockIdx.x, t = threadIdx.x;
    const float* L = logits + (size_t)r * VPAD;
    float mx = -3.4e38f;
    for (int i = t; i < VOCAB; i += 256) mx = fmaxf(mx, L[i]);
    red[t] = mx; __syncthreads();
    for (int s = 128; s > 0; s >>= 1) { if (t < s) red[t] = fmaxf(red[t], red[t + s]); __syncthreads(); }
    mx = red[0]; __syncthreads();
    float sm = 0.f;
    for (int i = t; i < VOCAB; i += 256) sm += __expf(L[i] - mx);
    red[t] = sm; __syncthreads();
    for (int s = 128; s > 0; s >>= 1) { if (t < s) red[t] += red[t + s]; __syncthreads(); }
    if (t == 0) { rowmax[r] = mx; rowlse[r] = __logf(red[0]); }
}

// ------------- phase-1 top-k: per-block top-16 over candidate slice --------
#define TOPK_BLOCKS 64
__global__ __launch_bounds__(256) void k_topk_part(
    const float* __restrict__ logits, const float* __restrict__ rowmax, const float* __restrict__ rowlse,
    const float* __restrict__ scores, const int* __restrict__ finished,
    int step, float* __restrict__ cv, int* __restrict__ ci) {
    __shared__ float sv[256 * 16];
    __shared__ int   si[256 * 16];
    __shared__ float rv[256];
    __shared__ int   rp[256];
    const int t = threadIdx.x, b = blockIdx.x;
    const int total = BEAM * VOCAB;
    const int chunk = (total + TOPK_BLOCKS - 1) / TOPK_BLOCKS;
    const int cstart = b * chunk;
    const int cend = (cstart + chunk < total) ? (cstart + chunk) : total;
    float lv[16]; int li[16];
#pragma unroll
    for (int i = 0; i < 16; ++i) { lv[i] = -3.4e38f; li[i] = 0; }
    for (int c = cstart + t; c < cend; c += 256) {
        int k = c / VOCAB, v = c - k * VOCAB;
        float tv;
        if (finished[k]) tv = scores[k] + ((v == EOS_TOK) ? 0.f : NEGF);
        else             tv = scores[k] + (logits[(size_t)k * VPAD + v] - rowmax[k] - rowlse[k]);
        if (step == 0 && k > 0) tv = NEGF;
        if (tv > lv[15]) {
            int p = 15;
            while (p > 0 && tv > lv[p - 1]) { lv[p] = lv[p - 1]; li[p] = li[p - 1]; --p; }
            lv[p] = tv; li[p] = c;
        }
    }
#pragma unroll
    for (int i = 0; i < 16; ++i) { sv[t * 16 + i] = lv[i]; si[t * 16 + i] = li[i]; }
    __syncthreads();
    for (int r = 0; r < 16; ++r) {
        float m = sv[t * 16]; int pos = t * 16;
#pragma unroll
        for (int i = 1; i < 16; ++i) { float x = sv[t * 16 + i]; if (x > m) { m = x; pos = t * 16 + i; } }
        rv[t] = m; rp[t] = pos; __syncthreads();
        for (int s = 128; s > 0; s >>= 1) {
            if (t < s && rv[t + s] > rv[t]) { rv[t] = rv[t + s]; rp[t] = rp[t + s]; }
            __syncthreads();
        }
        if (t == 0) { cv[b * 16 + r] = rv[0]; ci[b * 16 + r] = si[rp[0]]; sv[rp[0]] = -3.4e38f; }
        __syncthreads();
    }
}

// ---- phase-2 top-k: 1024 -> 16, then beam bookkeeping + state gather ------
__global__ __launch_bounds__(1024) void k_topk_final(
    const float* __restrict__ cv, const int* __restrict__ ci, int step,
    const float* __restrict__ h1n, const float* __restrict__ c1n,
    const float* __restrict__ h2n, const float* __restrict__ c2n, const float* __restrict__ ctxn,
    float* __restrict__ h1c, float* __restrict__ c1c,
    float* __restrict__ h2c, float* __restrict__ c2c, float* __restrict__ ctxc,
    float* __restrict__ scores, int* __restrict__ prev, int* __restrict__ finished,
    int* __restrict__ lengths, int* __restrict__ seqs) {
    __shared__ float scv[1024]; __shared__ int sci[1024];
    __shared__ float rv[1024];  __shared__ int rp[1024];
    __shared__ float selv[16];  __shared__ int seli[16];
    __shared__ int pb[16], nx[16];
    __shared__ int oldfin[16], oldlen[16];
    __shared__ int oldseq[BEAM * (MAXLEN + 1)];
    const int t = threadIdx.x;
    scv[t] = cv[t]; sci[t] = ci[t];
    if (t < 16) { oldfin[t] = finished[t]; oldlen[t] = lengths[t]; }
    for (int i = t; i < BEAM * (MAXLEN + 1); i += 1024) oldseq[i] = seqs[i];
    __syncthreads();
    for (int r = 0; r < 16; ++r) {
        rv[t] = scv[t]; rp[t] = t; __syncthreads();
        for (int s = 512; s > 0; s >>= 1) {
            if (t < s && rv[t + s] > rv[t]) { rv[t] = rv[t + s]; rp[t] = rp[t + s]; }
            __syncthreads();
        }
        if (t == 0) { selv[r] = rv[0]; seli[r] = sci[rp[0]]; scv[rp[0]] = -3.4e38f; }
        __syncthreads();
    }
    if (t < 16) {
        int idx = seli[t];
        int p = idx / VOCAB, n = idx - p * VOCAB;
        pb[t] = p; nx[t] = n;
        int wf = oldfin[p];
        lengths[t]  = wf ? oldlen[p] : (step + 2);
        finished[t] = (wf || n == EOS_TOK) ? 1 : 0;
        scores[t]   = selv[t];
        prev[t]     = n;
    }
    __syncthreads();
    for (int i = t; i < BEAM * (MAXLEN + 1); i += 1024) {
        int k = i / (MAXLEN + 1), pos = i - k * (MAXLEN + 1);
        seqs[i] = (pos == step + 1) ? nx[k] : oldseq[pb[k] * (MAXLEN + 1) + pos];
    }
    for (int i = t; i < BEAM * HID; i += 1024) {
        int k = i >> 9, j = i & 511;
        int src = pb[k] * HID + j;
        h1c[i] = h1n[src]; c1c[i] = c1n[src];
        h2c[i] = h2n[src]; c2c[i] = c2n[src];
        ctxc[i] = ctxn[src];
    }
}

// ------------------------------- final output ------------------------------
__global__ __launch_bounds__(256) void k_output(
    const int* __restrict__ seqs, const int* __restrict__ finished,
    const int* __restrict__ lengths, const float* __restrict__ scores, float* __restrict__ out) {
    const int t = threadIdx.x;
    for (int i = t; i < BEAM * (MAXLEN + 1); i += 256) {
        int k = i / (MAXLEN + 1), pos = i - k * (MAXLEN + 1);
        int tok = seqs[i];
        if (pos == MAXLEN && !finished[k]) tok = EOS_TOK;
        out[i] = (float)tok;
    }
    if (t < BEAM) out[BEAM * (MAXLEN + 1) + t] = scores[t] / powf((float)lengths[t], 1.2f);
}

// ===========================================================================
extern "C" void kernel_launch(void* const* d_in, const int* in_sizes, int n_in,
                              void* d_out, int out_size, void* d_ws, size_t ws_size,
                              hipStream_t stream) {
    (void)in_sizes; (void)n_in; (void)out_size; (void)ws_size;
    const float* enc_key   = (const float*)d_in[0];
    const float* enc_value = (const float*)d_in[1];
    const unsigned char* mask = (const unsigned char*)d_in[2];
    const float* embedding = (const float*)d_in[3];
    const float* W_ih1 = (const float*)d_in[4];
    const float* W_hh1 = (const float*)d_in[5];
    const float* b_ih1 = (const float*)d_in[6];
    const float* b_hh1 = (const float*)d_in[7];
    const float* W_ih2 = (const float*)d_in[8];
    const float* W_hh2 = (const float*)d_in[9];
    const float* b_ih2 = (const float*)d_in[10];
    const float* b_hh2 = (const float*)d_in[11];
    const float* Wq = (const float*)d_in[12];
    const float* bq = (const float*)d_in[13];
    const float* Wc = (const float*)d_in[14];
    const float* bc = (const float*)d_in[15];

    // -------- workspace bump allocator (256B aligned), deterministic -------
    char* ws = (char*)d_ws;
    size_t off = 0;
    auto alloc = [&](size_t bytes) -> char* {
        off = (off + 255) & ~(size_t)255;
        char* p = ws + off;
        off += bytes;
        return p;
    };
    bf16_t* B1    = (bf16_t*)alloc((size_t)2048 * 1536 * 2);   // [Wih1|Whh1]
    bf16_t* B2    = (bf16_t*)alloc((size_t)2048 * 1024 * 2);   // [Wih2|Whh2]
    bf16_t* BWq   = (bf16_t*)alloc((size_t)512 * 512 * 2);
    bf16_t* Bkey  = (bf16_t*)alloc((size_t)TSTEPS * QDIM * 2);
    bf16_t* BvalT = (bf16_t*)alloc((size_t)QDIM * TSTEPS * 2);
    bf16_t* BWc   = (bf16_t*)alloc((size_t)VOCAB * 1024 * 2);  // 103 MB, L2-resident
    float*  logits = (float*)alloc((size_t)BEAM * VPAD * 4);
    float*  energy = (float*)alloc((size_t)BEAM * TSTEPS * 4);
    bf16_t* attnbf = (bf16_t*)alloc((size_t)BEAM * TSTEPS * 2);
    float*  rowmax = (float*)alloc(BEAM * 4);
    float*  rowlse = (float*)alloc(BEAM * 4);
    float*  cand_v = (float*)alloc(TOPK_BLOCKS * 16 * 4);
    int*    cand_i = (int*)alloc(TOPK_BLOCKS * 16 * 4);
    float* h1c = (float*)alloc(BEAM * HID * 4);
    float* c1c = (float*)alloc(BEAM * HID * 4);
    float* h2c = (float*)alloc(BEAM * HID * 4);
    float* c2c = (float*)alloc(BEAM * HID * 4);
    float* ctxc = (float*)alloc(BEAM * QDIM * 4);
    float* h1n = (float*)alloc(BEAM * HID * 4);
    float* c1n = (float*)alloc(BEAM * HID * 4);
    float* h2n = (float*)alloc(BEAM * HID * 4);
    float* c2n = (float*)alloc(BEAM * HID * 4);
    float* ctxn = (float*)alloc(BEAM * QDIM * 4);
    float* scores = (float*)alloc(BEAM * 4);
    int* prevtok  = (int*)alloc(BEAM * 4);
    int* finished = (int*)alloc(BEAM * 4);
    int* lengths  = (int*)alloc(BEAM * 4);
    int* seqs     = (int*)alloc(BEAM * (MAXLEN + 1) * 4);

    // ---------------- one-time fp32 -> bf16 weight conversion --------------
    k_cvt_cat<<<512, 256, 0, stream>>>(W_ih1, W_hh1, B1, 2048, 1024, 512);
    k_cvt_cat<<<512, 256, 0, stream>>>(W_ih2, W_hh2, B2, 2048, 512, 512);
    k_cvt<<<256, 256, 0, stream>>>(Wq, BWq, (size_t)512 * 512);
    k_cvt<<<512, 256, 0, stream>>>(enc_key, Bkey, (size_t)TSTEPS * QDIM);
    k_cvt_valT<<<512, 256, 0, stream>>>(enc_value, BvalT);
    k_cvt<<<2048, 256, 0, stream>>>(Wc, BWc, (size_t)VOCAB * 1024);
    k_init<<<32, 256, 0, stream>>>(h1c, c1c, h2c, c2c, ctxc, scores, prevtok, finished, lengths, seqs);

    // ----------------------- 64 sequential decode steps --------------------
    for (int step = 0; step < MAXLEN; ++step) {
        k_lstm1<<<32, 128, 0, stream>>>(embedding, prevtok, ctxc, h1c, c1c,
                                        B1, b_ih1, b_hh1, h1n, c1n);
        k_lstm2<<<32, 128, 0, stream>>>(h1n, h2c, c2c, B2, b_ih2, b_hh2, h2n, c2n);
        k_attn_energy<<<16, 256, 0, stream>>>(h2n, BWq, bq, Bkey, energy);
        k_softmax<<<BEAM, 256, 0, stream>>>(energy, mask, attnbf);
        k_context<<<4, 256, 0, stream>>>(attnbf, BvalT, ctxn);
        k_logits<<<VTILES / 8, 256, 0, stream>>>(h2n, ctxn, BWc, bc, logits);
        k_rowstats<<<BEAM, 256, 0, stream>>>(logits, rowmax, rowlse);
        k_topk_part<<<TOPK_BLOCKS, 256, 0, stream>>>(logits, rowmax, rowlse, scores,
                                                     finished, step, cand_v, cand_i);
        k_topk_final<<<1, 1024, 0, stream>>>(cand_v, cand_i, step,
                                             h1n, c1n, h2n, c2n, ctxn,
                                             h1c, c1c, h2c, c2c, ctxc,
                                             scores, prevtok, finished, lengths, seqs);
    }
    k_output<<<1, 256, 0, stream>>>(seqs, finished, lengths, scores, (float*)d_out);
}
